// RNet_8555574854278
// MI455X (gfx1250) — matrix operand order repaired and checked by reading, not compiled, not run on hardware
//
#include <hip/hip_runtime.h>
#include <hip/hip_bf16.h>

// ---------------------------------------------------------------------------
// Types / helpers for CDNA5 WMMA (wave32, 16x16x32 bf16 -> f32)
// ---------------------------------------------------------------------------
typedef __attribute__((ext_vector_type(16))) __bf16 v16bf;
typedef __attribute__((ext_vector_type(8)))  float  v8f;

union Frag {
    v16bf v;
    unsigned short u[16];
    unsigned int   d[8];
};

__device__ __forceinline__ unsigned short f2bf(float f) {
    unsigned int u = __float_as_uint(f);
    u += 0x7FFFu + ((u >> 16) & 1u);      // round-to-nearest-even
    return (unsigned short)(u >> 16);
}
__device__ __forceinline__ float bf2f(unsigned short h) {
    return __uint_as_float(((unsigned int)h) << 16);
}

__device__ __forceinline__ v8f wmma_bf16(const Frag& a, const Frag& b, v8f c) {
    return __builtin_amdgcn_wmma_f32_16x16x32_bf16(
        false, a.v, false, b.v, (short)0, c, false, false);
}

// A fragment from a row-major matrix (global or LDS).
// p = &A[row][k0] + hi*8 (hi = lane>>4); element runs [0..7], [16..23].
__device__ __forceinline__ void load_a_rm(Frag& a, const unsigned short* p) {
    const uint4* q = (const uint4*)p;
    const uint4 x0 = q[0];
    const uint4 x1 = q[2];
    a.d[0] = x0.x; a.d[1] = x0.y; a.d[2] = x0.z; a.d[3] = x0.w;
    a.d[4] = x1.x; a.d[5] = x1.y; a.d[6] = x1.z; a.d[7] = x1.w;
}
// Pre-packed fragment: 16 bf16 contiguous per lane -> two b128 loads.
__device__ __forceinline__ void load_pk(Frag& b, const unsigned short* p) {
    const uint4* q = (const uint4*)p;
    const uint4 x0 = q[0];
    const uint4 x1 = q[1];
    b.d[0] = x0.x; b.d[1] = x0.y; b.d[2] = x0.z; b.d[3] = x0.w;
    b.d[4] = x1.x; b.d[5] = x1.y; b.d[6] = x1.z; b.d[7] = x1.w;
}

#define NN 196
#define NSQ 38416            // 196*196
#define WC_LD 224            // padded K for final GEMM (7*32)
#define WC_M 208             // padded M for final GEMM (13*16)

// ---------------------------------------------------------------------------
// f32 -> bf16 conversion (grid-stride)
// ---------------------------------------------------------------------------
__global__ void k_f32_to_bf16(const float* __restrict__ src,
                              unsigned short* __restrict__ dst, int n) {
    for (int i = blockIdx.x * blockDim.x + threadIdx.x; i < n;
         i += gridDim.x * blockDim.x)
        dst[i] = f2bf(src[i]);
}

// ---------------------------------------------------------------------------
// Pack a row-major f32 matrix W[K][N] into WMMA-B fragment layout:
// P[(kt*(N/16)+nt)*32 + lane][16 bf16]. One thread per (frag,lane).
// B layout: lanes 0-15 hold K k0..k0+7,k0+16..k0+23; lanes 16-31 hold k0+8..k0+15,k0+24..k0+31.
// ---------------------------------------------------------------------------
__global__ void k_pack_b(const float* __restrict__ W,
                         unsigned short* __restrict__ P, int K, int N) {
    const int total = (K >> 5) * (N >> 4) * 32;
    const int idx = blockIdx.x * 256 + threadIdx.x;
    if (idx >= total) return;
    const int lane = idx & 31, frag = idx >> 5;
    const int ntiles = N >> 4;
    const int kt = frag / ntiles, nt = frag % ntiles;
    const int hi = lane >> 4, nn = lane & 15;
    unsigned short* dst = P + (size_t)idx * 16;
    const int n = nt * 16 + nn;
#pragma unroll
    for (int v = 0; v < 8; ++v) {
        const int kA = kt * 32 + hi * 8 + 2 * v + (v >= 4 ? 8 : 0);
        dst[2 * v]     = f2bf(W[(size_t)kA * N + n]);
        dst[2 * v + 1] = f2bf(W[(size_t)(kA + 1) * N + n]);
    }
}

// ---------------------------------------------------------------------------
// Pack x (f32 [8][196][2048]) as the B operand of the final GEMM, with the
// K (=j) dimension zero-padded to 224. P[b][kt(7)][nt(128)][lane][16]
// ---------------------------------------------------------------------------
__global__ void k_pack_x(const float* __restrict__ x,
                         unsigned short* __restrict__ P) {
    const int idx = blockIdx.x * 256 + threadIdx.x;    // 8*7*128*32 = 229376
    if (idx >= 8 * 7 * 128 * 32) return;
    const int lane = idx & 31, frag = idx >> 5;
    const int nt = frag % 128;
    const int tmp = frag / 128;
    const int kt = tmp % 7, b = tmp / 7;
    const int hi = lane >> 4, nn = lane & 15;
    unsigned short* dst = P + (size_t)idx * 16;
    const float* xb_ = x + (size_t)b * NN * 2048;
    const int n = nt * 16 + nn;
#pragma unroll
    for (int v = 0; v < 8; ++v) {
        const int kA = kt * 32 + hi * 8 + 2 * v + (v >= 4 ? 8 : 0);
        dst[2 * v]     = (kA < NN)     ? f2bf(xb_[(size_t)kA * 2048 + n])       : (unsigned short)0;
        dst[2 * v + 1] = (kA + 1 < NN) ? f2bf(xb_[(size_t)(kA + 1) * 2048 + n]) : (unsigned short)0;
    }
}

// ---------------------------------------------------------------------------
// Projection: sub[1568,256] = x[1568,2048] @ w_prj + b_prj
// grid (98, 2), block 256 (8 waves); wave -> one 16x16 tile
// ---------------------------------------------------------------------------
__global__ void k_project(const unsigned short* __restrict__ xb,
                          const unsigned short* __restrict__ wp,   // packed
                          const float* __restrict__ bias,
                          unsigned short* __restrict__ sub) {
    const int wave = threadIdx.x >> 5, lane = threadIdx.x & 31;
    const int nn = lane & 15, hi = lane >> 4;
    const int mtile = blockIdx.x;
    const int ntile = blockIdx.y * 8 + wave;
    const int row = mtile * 16 + nn;
    const int col = ntile * 16 + nn;

    v8f acc;
    const float bv = bias[col];
    for (int r = 0; r < 8; ++r) acc[r] = bv;

    for (int kk = 0; kk < 64; ++kk) {
        Frag a, b;
        load_a_rm(a, xb + (size_t)row * 2048 + kk * 32 + hi * 8);
        load_pk(b, wp + ((size_t)(kk * 16 + ntile) * 32 + lane) * 16);
        acc = wmma_bf16(a, b, acc);
    }
#pragma unroll
    for (int r = 0; r < 8; ++r) {
        const int orow = mtile * 16 + r + 8 * hi;
        sub[(size_t)orow * 256 + col] = f2bf(acc[r]);
    }
}

// ---------------------------------------------------------------------------
// pair[i,j,c] = sub[i,c]*sub[j,c] for one batch (bf16, dword stores)
// grid 38416, block 128 (each thread: 2 channels)
// ---------------------------------------------------------------------------
__global__ void k_pair(const unsigned short* __restrict__ sub_b,
                       unsigned short* __restrict__ pair) {
    const int ij = blockIdx.x, c2 = threadIdx.x;     // c = 2*c2
    const int i = ij / NN, j = ij % NN;
    const unsigned int ua = *(const unsigned int*)(sub_b + i * 256 + 2 * c2);
    const unsigned int ub = *(const unsigned int*)(sub_b + j * 256 + 2 * c2);
    const float a0 = bf2f((unsigned short)ua), a1 = bf2f((unsigned short)(ua >> 16));
    const float b0 = bf2f((unsigned short)ub), b1 = bf2f((unsigned short)(ub >> 16));
    const unsigned int r = (unsigned int)f2bf(a0 * b0) |
                           ((unsigned int)f2bf(a1 * b1) << 16);
    *(unsigned int*)(pair + (size_t)ij * 256 + 2 * c2) = r;
}

// ---------------------------------------------------------------------------
// Pointwise GEMM + ReLU with M-blocking: each block computes MB m-tiles so
// every register-resident B fragment feeds MB WMMAs.
// grid ceil(2401/MB), block = (Cout/16)*32
// ---------------------------------------------------------------------------
template <int Cin, int Cout, int MB>
__global__ void k_pw(const unsigned short* __restrict__ in,
                     const unsigned short* __restrict__ wp,
                     const float* __restrict__ bias,
                     unsigned short* __restrict__ out) {
    constexpr int KST = Cin / 32, NT = Cout / 16;
    constexpr int ROWS = MB * 16;
    constexpr int QPR = Cin / 8;
    constexpr int QT = ROWS * QPR;
    __shared__ __align__(16) unsigned short smem[ROWS * Cin];

    const int tid = threadIdx.x;
    const int row0 = blockIdx.x * ROWS;

    for (int q = tid; q < QT; q += NT * 32) {
        const int r = q / QPR, qq = q % QPR;
        const int grow = row0 + r;
        uint4 val = {0u, 0u, 0u, 0u};
        if (grow < NSQ)
            val = *(const uint4*)(in + (size_t)grow * Cin + qq * 8);
        *(uint4*)(smem + r * Cin + qq * 8) = val;
    }
    __syncthreads();

    const int wave = tid >> 5, lane = tid & 31;
    const int nn = lane & 15, hi = lane >> 4;
    const int col = wave * 16 + nn;

    const float bv = bias[col];
    v8f acc[MB];
#pragma unroll
    for (int m = 0; m < MB; ++m)
        for (int r = 0; r < 8; ++r) acc[m][r] = bv;

#pragma unroll
    for (int kk = 0; kk < KST; ++kk) {
        Frag b;
        load_pk(b, wp + ((size_t)(kk * NT + wave) * 32 + lane) * 16);
#pragma unroll
        for (int m = 0; m < MB; ++m) {
            Frag a;
            load_a_rm(a, smem + (m * 16 + nn) * Cin + kk * 32 + hi * 8);
            acc[m] = wmma_bf16(a, b, acc[m]);
        }
    }
#pragma unroll
    for (int m = 0; m < MB; ++m) {
#pragma unroll
        for (int r = 0; r < 8; ++r) {
            const int orow = row0 + m * 16 + r + 8 * hi;
            if (orow < NSQ)
                out[(size_t)orow * Cout + col] = f2bf(fmaxf(acc[m][r], 0.f));
        }
    }
}

// ---------------------------------------------------------------------------
// 3x3 conv (dilated along j) + ReLU, M-blocked over MB consecutive output
// rows i: halo-padded (MB+2)x(16+2*DIL)xCin input staged in LDS with boundary
// zeros; each B fragment feeds MB WMMAs. Branch-free compute (EXEC all-1s).
// grid (ceil(196/MB), 13), block = (Cout/16)*32
// ---------------------------------------------------------------------------
template <int Cin, int Cout, int DIL, int MB>
__global__ void k_conv(const unsigned short* __restrict__ in,
                       const unsigned short* __restrict__ wp,
                       const float* __restrict__ bias,
                       unsigned short* __restrict__ out) {
    constexpr int KST = Cin / 32, NT = Cout / 16;
    constexpr int SR = MB + 2;                        // staged i rows
    constexpr int R  = 16 + 2 * DIL;                  // staged j positions
    constexpr int QPR = Cin / 8;
    constexpr int QT = SR * R * QPR;
    __shared__ __align__(16) unsigned short smem[SR * R * Cin];

    const int tid = threadIdx.x;
    const int i0 = blockIdx.x * MB;   // first output row of this block
    const int jt = blockIdx.y;        // j tile

    for (int q = tid; q < QT; q += NT * 32) {
        const int s = q / (R * QPR);
        const int rem = q % (R * QPR);
        const int r = rem / QPR, qq = rem % QPR;
        const int ii = i0 - 1 + s;
        const int jj = jt * 16 - DIL + r;
        uint4 val = {0u, 0u, 0u, 0u};
        if (ii >= 0 && ii < NN && jj >= 0 && jj < NN)
            val = *(const uint4*)(in + ((size_t)ii * NN + jj) * Cin + qq * 8);
        *(uint4*)(smem + ((s * R + r) * Cin) + qq * 8) = val;
    }
    __syncthreads();

    const int wave = tid >> 5, lane = tid & 31;
    const int nn = lane & 15, hi = lane >> 4;
    const int col = wave * 16 + nn;

    const float bv = bias[col];
    v8f acc[MB];
#pragma unroll
    for (int m = 0; m < MB; ++m)
        for (int r = 0; r < 8; ++r) acc[m][r] = bv;

    for (int tap = 0; tap < 9; ++tap) {
        const int di = tap / 3, t = tap % 3;
        const int jloc = nn + t * DIL;                // staged j column
        const unsigned short* wtap = wp + (size_t)tap * KST * NT * 32 * 16;
#pragma unroll
        for (int kk = 0; kk < KST; ++kk) {
            Frag b;
            load_pk(b, wtap + ((size_t)(kk * NT + wave) * 32 + lane) * 16);
#pragma unroll
            for (int m = 0; m < MB; ++m) {
                Frag a;
                load_a_rm(a, smem + ((size_t)((m + di) * R + jloc)) * Cin +
                                  kk * 32 + hi * 8);
                acc[m] = wmma_bf16(a, b, acc[m]);
            }
        }
    }
#pragma unroll
    for (int m = 0; m < MB; ++m) {
        const int io = i0 + m;
#pragma unroll
        for (int r = 0; r < 8; ++r) {
            const int jo = jt * 16 + r + 8 * hi;
            if (io < NN && jo < NN)
                out[((size_t)io * NN + jo) * Cout + col] =
                    f2bf(fmaxf(acc[m][r], 0.f));
        }
    }
}

// ---------------------------------------------------------------------------
// Cout=1 3x3 conv (VALU): logits[i,j] = relu(b + sum_{3x3x64} in*w)
// ---------------------------------------------------------------------------
__global__ void k_conv_n1(const unsigned short* __restrict__ in,
                          const float* __restrict__ w,
                          const float* __restrict__ bias,
                          float* __restrict__ out, int dil) {
    const int idx = blockIdx.x * blockDim.x + threadIdx.x;
    if (idx >= NSQ) return;
    const int i = idx / NN, j = idx % NN;
    float s = bias[0];
    for (int di = -1; di <= 1; ++di) {
        const int ii = i + di;
        if (ii < 0 || ii >= NN) continue;
        for (int t = 0; t < 3; ++t) {
            const int jj = j + (t - 1) * dil;
            if (jj < 0 || jj >= NN) continue;
            const unsigned short* p = in + ((size_t)ii * NN + jj) * 64;
            const float* wt = w + ((di + 1) * 3 + t) * 64;
#pragma unroll
            for (int c = 0; c < 64; ++c) s = fmaf(bf2f(p[c]), wt[c], s);
        }
    }
    out[idx] = fmaxf(s, 0.f);
}

// ---------------------------------------------------------------------------
// Cout=1 pointwise (VALU): logits0[idx] = relu(b + in[idx,:64] . w)
// ---------------------------------------------------------------------------
__global__ void k_pw_n1(const unsigned short* __restrict__ in,
                        const float* __restrict__ w,
                        const float* __restrict__ bias,
                        float* __restrict__ out) {
    const int idx = blockIdx.x * blockDim.x + threadIdx.x;
    if (idx >= NSQ) return;
    float s = bias[0];
    const unsigned short* p = in + (size_t)idx * 64;
#pragma unroll
    for (int c = 0; c < 64; ++c) s = fmaf(bf2f(p[c]), w[c], s);
    out[idx] = fmaxf(s, 0.f);
}

// ---------------------------------------------------------------------------
// Softmax stats over symmetrized logits: block bx = b*2 + branch
// ---------------------------------------------------------------------------
__global__ void k_softmax_stats(const float* __restrict__ L0,
                                const float* __restrict__ L1,
                                float* __restrict__ stats) {
    __shared__ float red[256];
    const int bx = blockIdx.x;
    const int b = bx >> 1;
    const float* L = ((bx & 1) ? L1 : L0) + (size_t)b * NSQ;

    float mx = -1e30f;
    for (int idx = threadIdx.x; idx < NSQ; idx += 256) {
        const int i = idx / NN, j = idx % NN;
        mx = fmaxf(mx, L[idx] + L[j * NN + i]);
    }
    red[threadIdx.x] = mx;
    __syncthreads();
    for (int s = 128; s > 0; s >>= 1) {
        if (threadIdx.x < s)
            red[threadIdx.x] = fmaxf(red[threadIdx.x], red[threadIdx.x + s]);
        __syncthreads();
    }
    mx = red[0];
    __syncthreads();

    float sum = 0.f;
    for (int idx = threadIdx.x; idx < NSQ; idx += 256) {
        const int i = idx / NN, j = idx % NN;
        sum += __expf(L[idx] + L[j * NN + i] - mx);
    }
    red[threadIdx.x] = sum;
    __syncthreads();
    for (int s = 128; s > 0; s >>= 1) {
        if (threadIdx.x < s) red[threadIdx.x] += red[threadIdx.x + s];
        __syncthreads();
    }
    if (threadIdx.x == 0) {
        stats[2 * bx]     = mx;
        stats[2 * bx + 1] = red[0];
    }
}

// ---------------------------------------------------------------------------
// Combined map (0.5*(softmax0+softmax1)) -> bf16, padded [8][208][224]
// ---------------------------------------------------------------------------
__global__ void k_combine(const float* __restrict__ L0,
                          const float* __restrict__ L1,
                          const float* __restrict__ stats,
                          unsigned short* __restrict__ Wc) {
    const int idx = blockIdx.x * 256 + threadIdx.x;
    if (idx >= 8 * WC_M * WC_LD) return;
    const int b = idx / (WC_M * WC_LD);
    const int rem = idx % (WC_M * WC_LD);
    const int i = rem / WC_LD, j = rem % WC_LD;
    float wv = 0.f;
    if (i < NN && j < NN) {
        const float* l0 = L0 + (size_t)b * NSQ;
        const float* l1 = L1 + (size_t)b * NSQ;
        const float s0 = l0[i * NN + j] + l0[j * NN + i];
        const float s1 = l1[i * NN + j] + l1[j * NN + i];
        const float m0 = stats[4 * b + 0], z0 = stats[4 * b + 1];
        const float m1 = stats[4 * b + 2], z1 = stats[4 * b + 3];
        wv = 0.5f * (__expf(s0 - m0) / z0 + __expf(s1 - m1) / z1);
    }
    Wc[idx] = f2bf(wv);
}

// ---------------------------------------------------------------------------
// Final: out[b,i,d] = sum_j Wc[b,i,j] * x[b,j,d]   (B operand pre-packed)
// grid (13, 16, 8), block 256
// ---------------------------------------------------------------------------
__global__ void k_final(const unsigned short* __restrict__ Wc,
                        const unsigned short* __restrict__ xp,
                        float* __restrict__ out) {
    const int wave = threadIdx.x >> 5, lane = threadIdx.x & 31;
    const int nn = lane & 15, hi = lane >> 4;
    const int mtile = blockIdx.x;
    const int ntile = blockIdx.y * 8 + wave;
    const int b = blockIdx.z;
    const int row = mtile * 16 + nn;        // i (< 208, zero-padded)
    const int col = ntile * 16 + nn;        // d
    const unsigned short* A  = Wc + (size_t)b * WC_M * WC_LD;
    const unsigned short* Bp = xp + (size_t)b * 7 * 128 * 32 * 16;

    v8f acc;
    for (int r = 0; r < 8; ++r) acc[r] = 0.f;

#pragma unroll
    for (int kk = 0; kk < 7; ++kk) {
        Frag a, bf;
        load_a_rm(a, A + (size_t)row * WC_LD + kk * 32 + hi * 8);
        load_pk(bf, Bp + ((size_t)(kk * 128 + ntile) * 32 + lane) * 16);
        acc = wmma_bf16(a, bf, acc);
    }
#pragma unroll
    for (int r = 0; r < 8; ++r) {
        const int i = mtile * 16 + r + 8 * hi;
        if (i < NN)
            out[((size_t)b * NN + i) * 2048 + col] = acc[r];
    }
}

// ---------------------------------------------------------------------------
// Host orchestration
// ---------------------------------------------------------------------------
extern "C" void kernel_launch(void* const* d_in, const int* in_sizes, int n_in,
                              void* d_out, int out_size, void* d_ws, size_t ws_size,
                              hipStream_t stream) {
    const float* x     = (const float*)d_in[0];
    const float* w_prj = (const float*)d_in[1];
    const float* b_prj = (const float*)d_in[2];
    const float* w01   = (const float*)d_in[3];
    const float* b01   = (const float*)d_in[4];
    const float* w02   = (const float*)d_in[5];
    const float* b02   = (const float*)d_in[6];
    const float* w03   = (const float*)d_in[7];
    const float* b03   = (const float*)d_in[8];
    const float* w1    = (const float*)d_in[9];
    const float* b1    = (const float*)d_in[10];
    const float* w2    = (const float*)d_in[11];
    const float* b2    = (const float*)d_in[12];
    const float* w3    = (const float*)d_in[13];
    const float* b3    = (const float*)d_in[14];

    char* ws = (char*)d_ws;
    size_t off = 0;
    auto alloc = [&](size_t bytes) -> void* {
        void* p = ws + off;
        off = (off + bytes + 255) & ~(size_t)255;
        return p;
    };
    unsigned short* xb     = (unsigned short*)alloc((size_t)1568 * 2048 * 2);
    unsigned short* wprj_p = (unsigned short*)alloc((size_t)64 * 16 * 32 * 16 * 2);
    unsigned short* w01_p  = (unsigned short*)alloc((size_t)8 * 8 * 32 * 16 * 2);
    unsigned short* w02_p  = (unsigned short*)alloc((size_t)4 * 4 * 32 * 16 * 2);
    unsigned short* w1_p   = (unsigned short*)alloc((size_t)72 * 8 * 32 * 16 * 2);
    unsigned short* w2_p   = (unsigned short*)alloc((size_t)36 * 4 * 32 * 16 * 2);
    unsigned short* xp     = (unsigned short*)alloc((size_t)8 * 7 * 128 * 32 * 16 * 2);
    unsigned short* subb   = (unsigned short*)alloc((size_t)1568 * 256 * 2);
    unsigned short* pairb  = (unsigned short*)alloc((size_t)NSQ * 256 * 2);
    unsigned short* Xa     = (unsigned short*)alloc((size_t)NSQ * 128 * 2);   // X0a / X1a
    unsigned short* Xb     = (unsigned short*)alloc((size_t)NSQ * 64 * 2);    // X0b / X1b
    float*          logit0 = (float*)alloc((size_t)8 * NSQ * 4);
    float*          logit1 = (float*)alloc((size_t)8 * NSQ * 4);
    float*          stats  = (float*)alloc(64 * 4);
    unsigned short* Wc     = (unsigned short*)alloc((size_t)8 * WC_M * WC_LD * 2);

    // x (A operand of projection) -> row-major bf16
    k_f32_to_bf16<<<dim3((1568 * 2048 + 255) / 256), 256, 0, stream>>>(
        x, xb, 1568 * 2048);

    // Pack all B operands into fragment layout (w1/w2 fold taps into K)
    auto packB = [&](const float* W, unsigned short* P, int K, int N) {
        const int tot = (K >> 5) * (N >> 4) * 32;
        k_pack_b<<<dim3((tot + 255) / 256), 256, 0, stream>>>(W, P, K, N);
    };
    packB(w_prj, wprj_p, 2048, 256);
    packB(w01,   w01_p,  256, 128);
    packB(w02,   w02_p,  128, 64);
    packB(w1,    w1_p,   9 * 256, 128);   // [tap*Cin][Cout] row-major
    packB(w2,    w2_p,   9 * 128, 64);
    k_pack_x<<<dim3((8 * 7 * 128 * 32 + 255) / 256), 256, 0, stream>>>(x, xp);

    k_project<<<dim3(98, 2), 256, 0, stream>>>(xb, wprj_p, b_prj, subb);

    for (int b = 0; b < 8; ++b) {
        const unsigned short* sub_b = subb + (size_t)b * NN * 256;
        k_pair<<<dim3(NSQ), 128, 0, stream>>>(sub_b, pairb);
        // branch 0 (pointwise chain)
        k_pw<256, 128, 6><<<dim3((2401 + 5) / 6), 256, 0, stream>>>(
            pairb, w01_p, b01, Xa);
        k_pw<128, 64, 8><<<dim3((2401 + 7) / 8), 128, 0, stream>>>(
            Xa, w02_p, b02, Xb);
        k_pw_n1<<<dim3((NSQ + 255) / 256), 256, 0, stream>>>(
            Xb, w03, b03, logit0 + (size_t)b * NSQ);
        // branch 1 (dilated conv chain) — reuses Xa/Xb
        k_conv<256, 128, 1, 4><<<dim3((NN + 3) / 4, 13), 256, 0, stream>>>(
            pairb, w1_p, b1, Xa);
        k_conv<128, 64, 2, 8><<<dim3((NN + 7) / 8, 13), 128, 0, stream>>>(
            Xa, w2_p, b2, Xb);
        k_conv_n1<<<dim3((NSQ + 255) / 256), 256, 0, stream>>>(
            Xb, w3, b3, logit1 + (size_t)b * NSQ, 4);
    }

    k_softmax_stats<<<dim3(16), 256, 0, stream>>>(logit0, logit1, stats);
    k_combine<<<dim3((8 * WC_M * WC_LD) / 256), 256, 0, stream>>>(
        logit0, logit1, stats, Wc);
    k_final<<<dim3(13, 16, 8), 256, 0, stream>>>(Wc, xp, (float*)d_out);

    (void)in_sizes; (void)n_in; (void)out_size; (void)ws_size;
}
